// RGCN2_7627861918259
// MI455X (gfx1250) — matrix operand order repaired and checked by reading, not compiled, not run on hardware
//
#include <hip/hip_runtime.h>
#include <hip/hip_bf16.h>

typedef __attribute__((ext_vector_type(16))) _Float16 v16h;
typedef __attribute__((ext_vector_type(8)))  float    v8f;
typedef unsigned int u32;
typedef unsigned short u16;

#define R_REL 16
#define F_IN  64
#define E_DIM 64
#define C_DIM 32
#define NBAS  16

union U16h { v16h v; u32 u[8]; };

static __device__ inline u32 pack2h(float a, float b) {
    union { _Float16 h[2]; u32 u; } p;
    p.h[0] = (_Float16)a;
    p.h[1] = (_Float16)b;
    return p.u;
}

// ---------------- meta / init kernels ----------------

__global__ void k_zero_meta(int* meta) {
    int i = threadIdx.x;
    if (i < 72) meta[i] = 0;
}

__global__ void k_zero_f32(float* p, int n) {
    for (int i = blockIdx.x * blockDim.x + threadIdx.x; i < n;
         i += gridDim.x * blockDim.x)
        p[i] = 0.f;
}

__global__ void k_init_out(float* out, const float* bias2, int n) {
    for (int i = blockIdx.x * blockDim.x + threadIdx.x; i < n * C_DIM;
         i += gridDim.x * blockDim.x)
        out[i] = bias2[i & (C_DIM - 1)];
}

__global__ void k_bias_relu(float* h1, const float* bias1, int n) {
    for (int i = blockIdx.x * blockDim.x + threadIdx.x; i < n * E_DIM;
         i += gridDim.x * blockDim.x) {
        float v = h1[i] + bias1[i & (E_DIM - 1)];
        h1[i] = v > 0.f ? v : 0.f;
    }
}

// ---------------- basis-composed weights (transposed, f16) ----------------
// w1t[r][e][f] = sum_b comps1[r,b] * bases1[b,f,e]   (f16)
// w2t[r][c][e] = sum_b comps2[r,b] * bases2[b,e,c]   (f16)
__global__ void k_weights(const float* __restrict__ comps1,
                          const float* __restrict__ bases1,
                          const float* __restrict__ comps2,
                          const float* __restrict__ bases2,
                          u16* __restrict__ w1t, u16* __restrict__ w2t) {
    int idx = blockIdx.x * blockDim.x + threadIdx.x;
    const int T1 = R_REL * E_DIM * F_IN;   // 65536
    const int T2 = R_REL * C_DIM * E_DIM;  // 32768
    if (idx < T1) {
        int r = idx >> 12;
        int e = (idx >> 6) & 63;
        int f = idx & 63;
        float s = 0.f;
        #pragma unroll
        for (int b = 0; b < NBAS; b++)
            s += comps1[r * NBAS + b] * bases1[(b * F_IN + f) * E_DIM + e];
        union { _Float16 h; u16 u; } c; c.h = (_Float16)s;
        w1t[(r * E_DIM + e) * F_IN + f] = c.u;
    } else if (idx < T1 + T2) {
        int j = idx - T1;
        int r = j >> 11;
        int c = (j >> 6) & 31;
        int e = j & 63;
        float s = 0.f;
        #pragma unroll
        for (int b = 0; b < NBAS; b++)
            s += comps2[r * NBAS + b] * bases2[(b * E_DIM + e) * C_DIM + c];
        union { _Float16 h; u16 u; } cv; cv.h = (_Float16)s;
        w2t[(r * C_DIM + c) * E_DIM + e] = cv.u;
    }
}

// ---------------- edge binning by relation ----------------

__global__ void k_hist(const int* __restrict__ rows, int nnz, int n,
                       int* __restrict__ counts) {
    __shared__ int lh[R_REL];
    if (threadIdx.x < R_REL) lh[threadIdx.x] = 0;
    __syncthreads();
    for (int i = blockIdx.x * blockDim.x + threadIdx.x; i < nnz;
         i += gridDim.x * blockDim.x)
        atomicAdd(&lh[rows[i] / n], 1);
    __syncthreads();
    if (threadIdx.x < R_REL) atomicAdd(&counts[threadIdx.x], lh[threadIdx.x]);
}

// meta layout (ints): [0..15]=counts [16..31]=bin_start [32..47]=cursor
//                     [48..64]=g_off (17)
__global__ void k_scan(int* meta) {
    if (threadIdx.x == 0 && blockIdx.x == 0) {
        int* counts    = meta;
        int* bin_start = meta + 16;
        int* cursor    = meta + 32;
        int* g_off     = meta + 48;
        int acc = 0, gacc = 0;
        for (int r = 0; r < R_REL; r++) {
            bin_start[r] = acc;
            cursor[r]    = acc;
            g_off[r]     = gacc;
            acc  += counts[r];
            gacc += (counts[r] + 15) >> 4;
        }
        g_off[R_REL] = gacc;
    }
}

__global__ void k_scatter(const int* __restrict__ rows,
                          const int* __restrict__ cols,
                          const float* __restrict__ vals,
                          int nnz, int n, int* meta,
                          int* __restrict__ col_s, int* __restrict__ row_s,
                          float* __restrict__ val_s) {
    __shared__ int lh[R_REL];
    __shared__ int lbase[R_REL];
    int* cursor = meta + 32;
    int chunk = (nnz + gridDim.x - 1) / gridDim.x;
    int s = blockIdx.x * chunk;
    int e = s + chunk; if (e > nnz) e = nnz;
    if (threadIdx.x < R_REL) lh[threadIdx.x] = 0;
    __syncthreads();
    for (int i = s + threadIdx.x; i < e; i += blockDim.x)
        atomicAdd(&lh[rows[i] / n], 1);
    __syncthreads();
    if (threadIdx.x < R_REL) {
        lbase[threadIdx.x] = atomicAdd(&cursor[threadIdx.x], lh[threadIdx.x]);
        lh[threadIdx.x] = 0;
    }
    __syncthreads();
    for (int i = s + threadIdx.x; i < e; i += blockDim.x) {
        int rv = rows[i];
        int r  = rv / n;
        int slot = lbase[r] + atomicAdd(&lh[r], 1);
        col_s[slot] = cols[i];
        row_s[slot] = rv - r * n;
        val_s[slot] = vals[i];
    }
}

// ---------------- fused (SpMM x W_r) WMMA + atomic scatter ----------------
// One wave per 16-edge group: A = diag(val) * X[cols]  (16 x 64, f16),
// B = w_r^T fragments (64 x EO), D scattered with f32 atomics into out.
// Tail-padded edges carry val=0/col=0/row=0, so their accumulator rows are
// exactly 0.0f and the scatter can run UNPREDICATED (atomic add of 0 is a
// no-op) -- no per-element exec manipulation.
template <int EB, int EO>
__global__ __launch_bounds__(256) void k_spmm_gemm(
    const float* __restrict__ src,       // [n, 64] f32 source features
    const u32*   __restrict__ wt,        // f16 pairs: [r][EO][64] halves
    const int*   __restrict__ meta,
    const int*   __restrict__ col_s,
    const int*   __restrict__ row_s,
    const float* __restrict__ val_s,
    float* __restrict__ out)             // [n, EO] f32, atomic accumulate
{
    // block-cached meta: [0..15]=counts [16..31]=bin_start [32..48]=g_off
    __shared__ int s_meta[65];
    __shared__ int   s_col[8][16];
    __shared__ int   s_row[8][16];
    __shared__ float s_val[8][16];

    if (threadIdx.x < 65)
        s_meta[threadIdx.x] =
            meta[threadIdx.x < 32 ? threadIdx.x : threadIdx.x + 16];
    __syncthreads();

    const int* lcnt  = s_meta;
    const int* lbin  = s_meta + 16;
    const int* lgoff = s_meta + 32;

    const int w      = threadIdx.x >> 5;
    const int l      = threadIdx.x & 31;
    const int m      = l & 15;
    const int half   = l >> 4;
    const int waveId = blockIdx.x * 8 + w;
    const int nWaves = gridDim.x * 8;
    const int g_tot  = lgoff[R_REL];

    int r_cur = -1;
    U16h bfrag[2][EB];

    for (int g = waveId; g < g_tot; g += nWaves) {
        // locate relation bin (LDS reads)
        int r = 0;
        while (r < R_REL - 1 && g >= lgoff[r + 1]) r++;
        int local = g - lgoff[r];
        int e0    = lbin[r] + local * 16;
        int valid = lcnt[r] - local * 16;
        if (valid > 16) valid = 16;

        // broadcast this group's 16 edges through per-wave LDS slots;
        // tail pad with zeros (makes everything downstream predicate-free)
        if (l < 16) {
            int ok = (l < valid);
            s_col[w][l] = ok ? col_s[e0 + l] : 0;
            s_row[w][l] = ok ? row_s[e0 + l] : 0;
            s_val[w][l] = ok ? val_s[e0 + l] : 0.f;
        }
        __builtin_amdgcn_wave_barrier();   // in-wave DS ops are in order

        // build A fragments (16x64 f16, K split lo/hi) per ISA A-layout:
        // VGPR v, lane l: k0 = 16*(v/4) + 2*(v&3) + 8*(l/16), M = l&15
        int   colm = s_col[w][m];
        float valm = s_val[w][m];
        const float* fr = src + colm * 64;
        U16h a_lo, a_hi;
        #pragma unroll
        for (int v = 0; v < 8; v++) {
            int k0 = ((v >> 2) << 4) + ((v & 3) << 1) + (half << 3);
            float2 x0 = *(const float2*)(fr + k0);
            float2 x1 = *(const float2*)(fr + k0 + 32);
            a_lo.u[v] = pack2h(x0.x * valm, x0.y * valm);
            a_hi.u[v] = pack2h(x1.x * valm, x1.y * valm);
        }

        // per-lane destination row offsets for the D scatter (reused by all
        // output tiles): lane l, VGPR v -> edge row M = v + 8*(l/16)
        int rowoff[8];
        #pragma unroll
        for (int v = 0; v < 8; v++)
            rowoff[v] = s_row[w][v + (half << 3)] * EO + m;

        // (re)load B fragments only when the bin changes (monotone in g)
        if (r != r_cur) {
            r_cur = r;
            #pragma unroll
            for (int kb = 0; kb < 2; kb++) {
                #pragma unroll
                for (int eb = 0; eb < EB; eb++) {
                    #pragma unroll
                    for (int v = 0; v < 8; v++) {
                        int k = kb * 32 + ((v >> 2) << 4) + ((v & 3) << 1) + (half << 3);
                        bfrag[kb][eb].u[v] =
                            wt[(r * EO + eb * 16 + m) * 32 + (k >> 1)];
                    }
                }
            }
        }

        // WMMA: D(16x16) per output block, K=64 via two chained x32 ops
        #pragma unroll
        for (int eb = 0; eb < EB; eb++) {
            v8f acc = {0.f, 0.f, 0.f, 0.f, 0.f, 0.f, 0.f, 0.f};
            acc = __builtin_amdgcn_wmma_f32_16x16x32_f16(
                false, a_lo.v, false, bfrag[0][eb].v, (short)0, acc, false, false);
            acc = __builtin_amdgcn_wmma_f32_16x16x32_f16(
                false, a_hi.v, false, bfrag[1][eb].v, (short)0, acc, false, false);
            // unpredicated scatter: padded rows contribute exact 0.0f
            #pragma unroll
            for (int v = 0; v < 8; v++)
                atomicAdd(out + rowoff[v] + eb * 16, acc[v]);
        }
    }
}

// ---------------- host-side launch ----------------

extern "C" void kernel_launch(void* const* d_in, const int* in_sizes, int n_in,
                              void* d_out, int out_size, void* d_ws, size_t ws_size,
                              hipStream_t stream) {
    const float* features  = (const float*)d_in[0];
    const int*   edge_rows = (const int*)  d_in[1];
    const int*   edge_cols = (const int*)  d_in[2];
    const float* edge_vals = (const float*)d_in[3];
    const float* comps1    = (const float*)d_in[4];
    const float* bases1    = (const float*)d_in[5];
    const float* comps2    = (const float*)d_in[6];
    const float* bases2    = (const float*)d_in[7];
    const float* bias1     = (const float*)d_in[8];
    const float* bias2     = (const float*)d_in[9];
    float* out = (float*)d_out;

    const int n   = in_sizes[0] / F_IN;   // 50000
    const int nnz = in_sizes[1];          // 1.6M

    // workspace carve-up (bytes)
    char* ws = (char*)d_ws;
    u16*   w1t   = (u16*)  (ws + 0);                       // 16*64*64 f16 = 128KB
    u16*   w2t   = (u16*)  (ws + 131072);                  // 16*32*64 f16 = 64KB
    int*   meta  = (int*)  (ws + 196608);                  // 72 ints (+pad)
    int*   col_s = (int*)  (ws + 197632);                  // nnz ints
    int*   row_s = (int*)  (ws + 197632 + (size_t)nnz * 4);
    float* val_s = (float*)(ws + 197632 + (size_t)nnz * 8);
    float* h1    = (float*)(ws + 197632 + (size_t)nnz * 12); // n*64 f32

    // 1. meta + weights
    k_zero_meta<<<1, 128, 0, stream>>>(meta);
    k_weights<<<(R_REL*E_DIM*F_IN + R_REL*C_DIM*E_DIM + 255) / 256, 256, 0, stream>>>(
        comps1, bases1, comps2, bases2, w1t, w2t);

    // 2. bin edges by relation (counting sort, LDS-aggregated)
    k_hist<<<1024, 256, 0, stream>>>(edge_rows, nnz, n, meta);
    k_scan<<<1, 32, 0, stream>>>(meta);
    k_scatter<<<1024, 256, 0, stream>>>(edge_rows, edge_cols, edge_vals,
                                        nnz, n, meta, col_s, row_s, val_s);

    // 3. init accumulators
    k_zero_f32<<<2048, 256, 0, stream>>>(h1, n * E_DIM);
    k_init_out<<<2048, 256, 0, stream>>>(out, bias2, n);

    // 4. layer 1: h1 = spmm(X) @ w1_r  (fused, WMMA + f32 atomics)
    k_spmm_gemm<4, 64><<<512, 256, 0, stream>>>(features, (const u32*)w1t, meta,
                                                col_s, row_s, val_s, h1);
    // 5. bias + relu
    k_bias_relu<<<2048, 256, 0, stream>>>(h1, bias1, n);

    // 6. layer 2: out += spmm(h1) @ w2_r
    k_spmm_gemm<2, 32><<<512, 256, 0, stream>>>(h1, (const u32*)w2t, meta,
                                                col_s, row_s, val_s, out);
}
